// MemorizingTransformer_72129680769663
// MI455X (gfx1250) — matrix operand order repaired and checked by reading, not compiled, not run on hardware
//
#include <hip/hip_runtime.h>
#include <math.h>

// ---------------------------------------------------------------------------
// CDNA5 (gfx1250) WMMA types & helpers
// ---------------------------------------------------------------------------
typedef __bf16 bf16_t;
typedef bf16_t v16bf __attribute__((ext_vector_type(16)));
typedef bf16_t v4bf  __attribute__((ext_vector_type(4)));
typedef float  v8f   __attribute__((ext_vector_type(8)));
typedef float  v4f   __attribute__((ext_vector_type(4)));
typedef int    v4i   __attribute__((ext_vector_type(4)));

union FragB16 { unsigned short u[16]; v16bf v; };

#if defined(__AMDGCN__) && __has_builtin(__builtin_amdgcn_global_load_async_to_lds_b128)
#define ASYNC_LDS_AVAILABLE 1
#else
#define ASYNC_LDS_AVAILABLE 0
#endif

__device__ __forceinline__ v8f v8f_zero() {
  v8f z;
#pragma unroll
  for (int i = 0; i < 8; ++i) z[i] = 0.0f;
  return z;
}

// Hardware f32->bf16 (RNE) conversions (lowers to v_cvt_pk_bf16_f32).
__device__ __forceinline__ void store_bf16(unsigned short* dst, float x) {
  *(bf16_t*)dst = (bf16_t)x;
}
__device__ __forceinline__ void pack4(unsigned short* dst, float4 v) {
  v4f f = { v.x, v.y, v.z, v.w };
  *(v4bf*)dst = __builtin_convertvector(f, v4bf);
}

// Fragment-major position of A-tile element (row 0..15, k 0..31):
// lanes 0-15 hold K=0..15, lanes 16-31 hold K=16..31.
__device__ __forceinline__ int frag_a_pos(int row, int k) {
  const int lane = row + ((k >= 16) ? 16 : 0);
  return lane * 16 + (k & 15);
}

// Fragment-major position of B-tile element (k 0..31, n 0..15):
// lanes 0-15 hold K=0..15, lanes 16-31 hold K=16..31; N = lane&15.
__device__ __forceinline__ int frag_b_pos(int k, int n) {
  const int lane = n + ((k >= 16) ? 16 : 0);
  return lane * 16 + (k & 15);
}

// One 32-byte contiguous LDS read per lane -> 2x ds_load_b128.
__device__ __forceinline__ v16bf load_frag(const unsigned short* tile, int lane) {
  return *(const v16bf*)(tile + lane * 16);
}

#define EPI_BIAS  1
#define EPI_RESID 2
#define EPI_GELU  4

// ---------------------------------------------------------------------------
// Generic GEMM: C = epi(A @ W)   A:MxK f32, W:KxN f32, C:MxN f32
// N, K compile-time -> strided W loads become immediate-offset loads and all
// address math strength-reduces. Assumes M%64==0, N%64==0, K%32==0.
// 64x64 block tile, 8 wave32 waves, each wave = 16x32 via two WMMA tiles.
// Double-buffered fragment-major LDS tiles; one barrier per K-step.
// ---------------------------------------------------------------------------
template<int N, int K>
__device__ __forceinline__ void stage_tiles(const float* __restrict__ A,
                                            const float* __restrict__ W,
                                            unsigned short* __restrict__ As,
                                            unsigned short* __restrict__ Bs,
                                            int tid, int bm, int bn, int kk)
{
#pragma unroll
  for (int it = 0; it < 2; ++it) {
    const int g  = tid + it * 256;        // 0..511
    const int r  = g >> 3;                // 0..63
    const int c4 = (g & 7) * 4;           // 0..28
    float4 va = *(const float4*)&A[(size_t)(bm + r) * K + kk + c4];
    pack4(&As[(r >> 4) * 512 + frag_a_pos(r & 15, c4)], va);
  }
#pragma unroll
  for (int it = 0; it < 2; ++it) {
    const int g  = tid + it * 256;        // 0..511
    const int n  = g & 63;
    const int kq = (g >> 6) * 4;          // 0..28
    const float* wb = &W[(size_t)(kk + kq) * N + bn + n];
    float4 vb;                             // constant strides -> inst offsets
    vb.x = wb[0];
    vb.y = wb[N];
    vb.z = wb[2 * N];
    vb.w = wb[3 * N];
    pack4(&Bs[(n >> 4) * 512 + frag_b_pos(kq, n & 15)], vb);
  }
}

template<int N, int K, int EPI>
__global__ __launch_bounds__(256)
void gemm_bf16_wmma(const float* __restrict__ A, const float* __restrict__ W,
                    const float* __restrict__ bias, const float* __restrict__ resid,
                    float* __restrict__ C)
{
  __shared__ __align__(32) unsigned short As[2 * 2048];  // 2 bufs x 4 row tiles
  __shared__ __align__(32) unsigned short Bs[2 * 2048];  // 2 bufs x 4 col tiles
  const int tid  = threadIdx.x;
  const int lane = tid & 31;
  const int wid  = tid >> 5;
  const int bm = blockIdx.y * 64;
  const int bn = blockIdx.x * 64;
  const int wm  = wid & 3;            // A row-tile index
  const int wnt = (wid >> 2) * 2;     // first of two B col-tiles

  v8f acc0 = v8f_zero(), acc1 = v8f_zero();

  stage_tiles<N, K>(A, W, As, Bs, tid, bm, bn, 0);
  int buf = 0;
  for (int kk = 0; kk < K; kk += 32, buf ^= 1) {
    __syncthreads();
    if (kk + 32 < K)
      stage_tiles<N, K>(A, W, As + (buf ^ 1) * 2048, Bs + (buf ^ 1) * 2048,
                        tid, bm, bn, kk + 32);
    if (kk + 64 < K) {                   // global_prefetch_b8 for tile after next
      __builtin_prefetch(&A[(size_t)(bm + (tid >> 2)) * K + kk + 64], 0, 1);
      __builtin_prefetch(&W[(size_t)(kk + 64 + (tid >> 3)) * N + bn], 0, 1);
    }
    v16bf a  = load_frag(&As[buf * 2048 + wm * 512], lane);
    v16bf b0 = load_frag(&Bs[buf * 2048 + wnt * 512], lane);
    v16bf b1 = load_frag(&Bs[buf * 2048 + (wnt + 1) * 512], lane);
    acc0 = __builtin_amdgcn_wmma_f32_16x16x32_bf16(false, a, false, b0, (short)0, acc0, false, false);
    acc1 = __builtin_amdgcn_wmma_f32_16x16x32_bf16(false, a, false, b1, (short)0, acc1, false, false);
  }

  const int col = lane & 15;
  const int rb  = (lane >> 4) ? 8 : 0;
#pragma unroll
  for (int t = 0; t < 2; ++t) {
    const int c = bn + (wnt + t) * 16 + col;
#pragma unroll
    for (int v = 0; v < 8; ++v) {
      const int r = bm + wm * 16 + rb + v;
      float val = (t == 0) ? acc0[v] : acc1[v];
      if (EPI & EPI_BIAS)  val += bias[c];
      if (EPI & EPI_GELU)  val = 0.5f * val * (1.0f + erff(val * 0.70710678118654752f));
      if (EPI & EPI_RESID) val += resid[(size_t)r * N + c];
      C[(size_t)r * N + c] = val;
    }
  }
}

// ---------------------------------------------------------------------------
// Embedding gather (float4 granularity)
// ---------------------------------------------------------------------------
__global__ __launch_bounds__(256)
void embed_kernel(const int* __restrict__ tokens, const float* __restrict__ emb,
                  float* __restrict__ x, int rows, int D)
{
  size_t idx = (size_t)blockIdx.x * blockDim.x + threadIdx.x;   // float4 index
  if (idx >= (size_t)rows * (D / 4)) return;
  const int r = (int)(idx / (D / 4)), c4 = (int)(idx % (D / 4)) * 4;
  *(float4*)&x[(size_t)r * D + c4] = *(const float4*)&emb[(size_t)tokens[r] * D + c4];
}

// ---------------------------------------------------------------------------
// LayerNorm (one 256-thread block per row)
// ---------------------------------------------------------------------------
__global__ __launch_bounds__(256)
void ln_kernel(const float* __restrict__ x, const float* __restrict__ w,
               const float* __restrict__ b, float* __restrict__ y, int D)
{
  __shared__ float red[256];
  const int row = blockIdx.x;
  const int tid = threadIdx.x;
  const float* xr = x + (size_t)row * D;
  float s = 0.0f;
  for (int i = tid; i < D; i += 256) s += xr[i];
  red[tid] = s;
  __syncthreads();
  for (int o = 128; o > 0; o >>= 1) { if (tid < o) red[tid] += red[tid + o]; __syncthreads(); }
  const float mean = red[0] / (float)D;
  __syncthreads();
  float v = 0.0f;
  for (int i = tid; i < D; i += 256) { float d = xr[i] - mean; v += d * d; }
  red[tid] = v;
  __syncthreads();
  for (int o = 128; o > 0; o >>= 1) { if (tid < o) red[tid] += red[tid + o]; __syncthreads(); }
  const float inv = rsqrtf(red[0] / (float)D + 1e-5f);
  for (int i = tid; i < D; i += 256)
    y[(size_t)row * D + i] = (xr[i] - mean) * inv * w[i] + b[i];
}

// ---------------------------------------------------------------------------
// Build concatenated K ([XL mems | new tokens], l2-normalized new rows on the
// kNN layer, device-side branch) and transposed V (vt[b][d][j]).
// ---------------------------------------------------------------------------
__global__ __launch_bounds__(256)
void build_kv_kernel(const float* __restrict__ kv, const float* __restrict__ xl,
                     const int* __restrict__ knn_layer_p, float* __restrict__ kc,
                     float* __restrict__ vt,
                     int layer, int B, int N, int XL)
{
  const int J = XL + N;
  const int r = blockIdx.x * blockDim.x + threadIdx.x;
  if (r >= B * J) return;
  const int b = r / J, j = r % J;
  const bool norm_new = (*knn_layer_p == layer);
  float kr[64], vr[64];
  if (j < XL) {
    const float* base = xl + ((((size_t)layer * B + b) * XL + j) * 2) * 64;
#pragma unroll
    for (int d = 0; d < 64; ++d) { kr[d] = base[d]; vr[d] = base[64 + d]; }
  } else {
    const float* base = kv + ((size_t)b * N + (j - XL)) * 128;
#pragma unroll
    for (int d = 0; d < 64; ++d) { kr[d] = base[d]; vr[d] = base[64 + d]; }
    if (norm_new) {
      float ss = 0.0f;
#pragma unroll
      for (int d = 0; d < 64; ++d) ss += kr[d] * kr[d];
      const float inv = 1.0f / fmaxf(sqrtf(ss), 1e-12f);
#pragma unroll
      for (int d = 0; d < 64; ++d) kr[d] *= inv;
    }
  }
#pragma unroll
  for (int d = 0; d < 64; ++d) {
    kc[(size_t)r * 64 + d] = kr[d];
    vt[((size_t)b * 64 + d) * J + j] = vr[d];
  }
}

// ---------------------------------------------------------------------------
// Normalize db keys (one thread per (b,m))
// ---------------------------------------------------------------------------
__global__ __launch_bounds__(256)
void normalize_db_kernel(const float* __restrict__ db, float* __restrict__ out, int rows)
{
  const int r = blockIdx.x * blockDim.x + threadIdx.x;
  if (r >= rows) return;
  const float4* kp = (const float4*)(db + (size_t)r * 2 * 64);
  float4 kv4[16];
  float ss = 0.0f;
#pragma unroll
  for (int q = 0; q < 16; ++q) {
    kv4[q] = kp[q];
    ss += kv4[q].x * kv4[q].x + kv4[q].y * kv4[q].y + kv4[q].z * kv4[q].z + kv4[q].w * kv4[q].w;
  }
  const float inv = 1.0f / fmaxf(sqrtf(ss), 1e-12f);
  float4* op = (float4*)(out + (size_t)r * 64);
#pragma unroll
  for (int q = 0; q < 16; ++q) {
    float4 t = kv4[q];
    t.x *= inv; t.y *= inv; t.z *= inv; t.w *= inv;
    op[q] = t;
  }
}

// ---------------------------------------------------------------------------
// Attention: one wave32 per 16-query block of one (b,h).
// sim = q@k^T (WMMA) -> mask/scale -> [kNN top-32 over db] -> softmax ->
// out = probs@v (WMMA, v transposed) + mem-value contribution.
// ---------------------------------------------------------------------------
__global__ __launch_bounds__(32)
void attention_wmma(const float* __restrict__ qbuf, const float* __restrict__ kcat,
                    const float* __restrict__ vt, const float* __restrict__ dbkn,
                    const float* __restrict__ db_kv, const float* __restrict__ knn_scale,
                    const int* __restrict__ knn_layer_p, const int* __restrict__ topk_p,
                    float* __restrict__ out,
                    int layer, int B, int H, int N, int J, int XL, int M)
{
  extern __shared__ __align__(32) char smraw[];
  size_t off = 0;
  float* qs   = (float*)(smraw + off); off += 16 * 64 * sizeof(float);
  float* sim  = (float*)(smraw + off); off += (size_t)16 * J * sizeof(float);
  float* smv  = (float*)(smraw + off); off += 16 * 32 * sizeof(float);
  float* outm = (float*)(smraw + off); off += 16 * 64 * sizeof(float);
  int*   sidx = (int*)  (smraw + off); off += 16 * 32 * sizeof(int);
  unsigned short* qbs = (unsigned short*)(smraw + off); off += 2 * 512 * sizeof(unsigned short);
  unsigned short* pbs = (unsigned short*)(smraw + off);   // (J/32) tiles * 512

  const int lane = threadIdx.x & 31;
  const int nb = N / 16;
  const int ib = blockIdx.x % nb;
  const int bh = blockIdx.x / nb;
  const int h = bh % H;
  const int b = bh / H;
  const int i0 = ib * 16;

  const bool is_knn = (*knn_layer_p == layer);
  int topk = *topk_p; if (topk > 32) topk = 32; if (topk < 1) topk = 1;
  const float scale = is_knn ? __expf(knn_scale[h]) : 1.0f;
  const float NEGF = 3.402823466e38f;

  // ---- load q tile into LDS (async DMA when available) ----
#if ASYNC_LDS_AVAILABLE
  for (int g = lane; g < 16 * 16; g += 32) {
    const int r = g >> 4, d4 = (g & 15) * 4;
    const float* src = &qbuf[((size_t)(b * N + i0 + r)) * (H * 64) + h * 64 + d4];
    __builtin_amdgcn_global_load_async_to_lds_b128(
        (v4i*)src, (v4i*)&qs[r * 64 + d4], 0, 0);
  }
#if __has_builtin(__builtin_amdgcn_s_wait_asynccnt)
  __builtin_amdgcn_s_wait_asynccnt(0);
#else
  asm volatile("s_wait_asynccnt 0x0" ::: "memory");
#endif
#else
  for (int g = lane; g < 16 * 16; g += 32) {
    const int r = g >> 4, d4 = (g & 15) * 4;
    *(float4*)&qs[r * 64 + d4] =
        *(const float4*)&qbuf[((size_t)(b * N + i0 + r)) * (H * 64) + h * 64 + d4];
  }
#endif
  __syncthreads();
  if (lane < 16) {
    float ss = 0.0f;
#pragma unroll
    for (int d = 0; d < 64; ++d) { const float t = qs[lane * 64 + d]; ss += t * t; }
    const float inv = is_knn ? (1.0f / fmaxf(sqrtf(ss), 1e-12f)) : rsqrtf(64.0f);
#pragma unroll
    for (int d = 0; d < 64; ++d) {
      const float t = qs[lane * 64 + d] * inv;
      qs[lane * 64 + d] = t;
      store_bf16(&qbs[(d >> 5) * 512 + frag_a_pos(lane, d & 31)], t);
    }
  }
  __syncthreads();

  const v16bf aq0 = load_frag(qbs, lane);        // K = 0..31
  const v16bf aq1 = load_frag(qbs + 512, lane);  // K = 32..63

  const int col = lane & 15;
  const int kb  = (lane >> 4) ? 16 : 0;
  const int rbv = (lane >> 4) ? 8 : 0;

  // ---- sim = q @ k^T over all J columns, 16 at a time ----
  for (int jt = 0; jt < J; jt += 16) {
    FragB16 fb0, fb1;
    const float4* kp4 = (const float4*)(kcat + (size_t)(b * J + jt + col) * 64 + kb);
    pack4(&fb0.u[0],  kp4[0]);  pack4(&fb0.u[4],  kp4[1]);
    pack4(&fb0.u[8],  kp4[2]);  pack4(&fb0.u[12], kp4[3]);
    pack4(&fb1.u[0],  kp4[8]);  pack4(&fb1.u[4],  kp4[9]);
    pack4(&fb1.u[8],  kp4[10]); pack4(&fb1.u[12], kp4[11]);
    v8f acc = v8f_zero();
    acc = __builtin_amdgcn_wmma_f32_16x16x32_bf16(false, aq0, false, fb0.v, (short)0, acc, false, false);
    acc = __builtin_amdgcn_wmma_f32_16x16x32_bf16(false, aq1, false, fb1.v, (short)0, acc, false, false);
    const int jj = jt + col;
#pragma unroll
    for (int v = 0; v < 8; ++v) {
      const int i = i0 + rbv + v;
      float s = acc[v] * scale;
      if (jj > i + XL) s = -NEGF;            // causal mask
      sim[(rbv + v) * J + jj] = s;
    }
  }
  __syncthreads();

  // ---- kNN retrieval: per-row top-32 over M normalized db keys ----
  if (is_knn && lane < 16) {
    float bv[32]; int bi[32];
#pragma unroll
    for (int t = 0; t < 32; ++t) { bv[t] = -NEGF; bi[t] = 0; }
    const float4* qr4 = (const float4*)(qs + lane * 64);
    float4 q4[16];
#pragma unroll
    for (int qd = 0; qd < 16; ++qd) q4[qd] = qr4[qd];
    for (int m = 0; m < M; ++m) {
      const float4* km4 = (const float4*)(dbkn + (size_t)(b * M + m) * 64);
      float s = 0.0f;
#pragma unroll
      for (int qd = 0; qd < 16; ++qd) {
        const float4 kmv = km4[qd];
        s += q4[qd].x * kmv.x + q4[qd].y * kmv.y + q4[qd].z * kmv.z + q4[qd].w * kmv.w;
      }
      if (s > bv[31]) {
        int p = 31;
        while (p > 0 && bv[p - 1] < s) { bv[p] = bv[p - 1]; bi[p] = bi[p - 1]; --p; }
        bv[p] = s; bi[p] = m;
      }
    }
    for (int t = 0; t < 32; ++t) {
      smv[lane * 32 + t]  = bv[t] * scale;
      sidx[lane * 32 + t] = bi[t];
    }
  }
  __syncthreads();

  // ---- joint softmax over [mem (topk) | local (J)] + mem-value accumulation ----
  if (lane < 16) {
    float mx = -NEGF;
    if (is_knn) for (int t = 0; t < topk; ++t) mx = fmaxf(mx, smv[lane * 32 + t]);
    for (int j = 0; j < J; ++j) mx = fmaxf(mx, sim[lane * J + j]);
    float sum = 0.0f;
    if (is_knn) {
      for (int t = 0; t < topk; ++t) {
        const float e = __expf(smv[lane * 32 + t] - mx);
        smv[lane * 32 + t] = e; sum += e;
      }
    }
    for (int j = 0; j < J; ++j) {
      const float e = __expf(sim[lane * J + j] - mx);
      sim[lane * J + j] = e; sum += e;
    }
    const float inv = 1.0f / sum;
    for (int j = 0; j < J; ++j)
      store_bf16(&pbs[(j >> 5) * 512 + frag_a_pos(lane, j & 31)], sim[lane * J + j] * inv);
#pragma unroll
    for (int d = 0; d < 64; ++d) outm[lane * 64 + d] = 0.0f;
    if (is_knn) {
      for (int t = 0; t < topk; ++t) {
        const float w = smv[lane * 32 + t] * inv;
        const float* vm = db_kv + (((size_t)(b * M + sidx[lane * 32 + t]) * 2) + 1) * 64;
#pragma unroll
        for (int d = 0; d < 64; ++d) outm[lane * 64 + d] += w * vm[d];
      }
    }
  }
  __syncthreads();

  // ---- out = probs @ v  via WMMA; B fragments from transposed v (contiguous) ----
  v8f oacc[4];
#pragma unroll
  for (int t = 0; t < 4; ++t) oacc[t] = v8f_zero();
  for (int kt = 0; kt < J; kt += 32) {
    const v16bf ap = load_frag(&pbs[(kt >> 5) * 512], lane);
#pragma unroll
    for (int t = 0; t < 4; ++t) {
      FragB16 fb;
      const float4* vp4 = (const float4*)(vt + (size_t)(b * 64 + t * 16 + col) * J + kt + kb);
      pack4(&fb.u[0], vp4[0]); pack4(&fb.u[4],  vp4[1]);
      pack4(&fb.u[8], vp4[2]); pack4(&fb.u[12], vp4[3]);
      oacc[t] = __builtin_amdgcn_wmma_f32_16x16x32_bf16(false, ap, false, fb.v, (short)0, oacc[t], false, false);
    }
  }
#pragma unroll
  for (int t = 0; t < 4; ++t) {
#pragma unroll
    for (int v = 0; v < 8; ++v) {
      const int r = rbv + v;
      const int d = t * 16 + col;
      out[((size_t)(b * N + i0 + r)) * (H * 64) + h * 64 + d] = oacc[t][v] + outm[r * 64 + d];
    }
  }
}

// ---------------------------------------------------------------------------
// Host orchestration
// ---------------------------------------------------------------------------
extern "C" void kernel_launch(void* const* d_in, const int* in_sizes, int n_in,
                              void* d_out, int out_size, void* d_ws, size_t ws_size,
                              hipStream_t stream)
{
  (void)in_sizes; (void)n_in; (void)out_size; (void)ws_size;
  const int B = 2, N = 1024, DIM = 1024, H = 16, DH = 64, DEPTH = 4;
  const int V = 32000, XL = 256, M = 4096;
  const int J = XL + N;
  const int ROWS = B * N;

  const int*   tokens    = (const int*)d_in[0];
  const float* emb       = (const float*)d_in[1];
  const float* ln1_w     = (const float*)d_in[2];
  const float* ln1_b     = (const float*)d_in[3];
  const float* wq        = (const float*)d_in[4];
  const float* wkv       = (const float*)d_in[5];
  const float* wo        = (const float*)d_in[6];
  const float* wo_b      = (const float*)d_in[7];
  const float* knn_scale = (const float*)d_in[8];
  const float* ln2_w     = (const float*)d_in[9];
  const float* ln2_b     = (const float*)d_in[10];
  const float* ff_w1     = (const float*)d_in[11];
  const float* ff_b1     = (const float*)d_in[12];
  const float* ff_w2     = (const float*)d_in[13];
  const float* ff_b2     = (const float*)d_in[14];
  const float* lnf_w     = (const float*)d_in[15];
  const float* lnf_b     = (const float*)d_in[16];
  const float* w_logits  = (const float*)d_in[17];
  const float* b_logits  = (const float*)d_in[18];
  const float* xl_mems   = (const float*)d_in[19];
  const float* db_kv     = (const float*)d_in[20];
  const int*   knn_layer = (const int*)d_in[21];
  const int*   topk      = (const int*)d_in[22];

  char* p = (char*)d_ws;
  auto carve = [&](size_t bytes) -> char* {
    char* r = p; p += (bytes + 255) & ~(size_t)255; return r;
  };
  float* x    = (float*)carve((size_t)ROWS * DIM * 4);
  float* x2   = (float*)carve((size_t)ROWS * DIM * 4);
  float* hb   = (float*)carve((size_t)ROWS * DIM * 4);
  float* qb   = (float*)carve((size_t)ROWS * H * DH * 4);
  float* kvb  = (float*)carve((size_t)ROWS * 2 * DH * 4);
  float* kc   = (float*)carve((size_t)B * J * DH * 4);
  float* vt   = (float*)carve((size_t)B * DH * J * 4);
  float* ao   = (float*)carve((size_t)ROWS * H * DH * 4);
  float* ffh  = (float*)carve((size_t)ROWS * 4 * DIM * 4);
  float* dbkn = (float*)carve((size_t)B * M * DH * 4);

  dim3 blk256(256), blk32(32);

  {
    size_t tot = (size_t)ROWS * (DIM / 4);
    embed_kernel<<<(unsigned)((tot + 255) / 256), blk256, 0, stream>>>(tokens, emb, x, ROWS, DIM);
  }
  normalize_db_kernel<<<(B * M + 255) / 256, blk256, 0, stream>>>(db_kv, dbkn, B * M);

  const size_t attn_shmem =
      (size_t)16 * 64 * 4 + (size_t)16 * J * 4 + 16 * 32 * 4 + 16 * 64 * 4 +
      16 * 32 * 4 + 2 * 512 * 2 + (size_t)(J / 32) * 512 * 2;

  const dim3 gDIM(DIM / 64, ROWS / 64);        // N=1024 outputs
  const dim3 gKV(2 * DH / 64, ROWS / 64);      // N=128 outputs
  const dim3 gFF(4 * DIM / 64, ROWS / 64);     // N=4096 outputs
  const dim3 gLG(V / 64, ROWS / 64);           // N=32000 outputs

  for (int l = 0; l < DEPTH; ++l) {
    ln_kernel<<<ROWS, blk256, 0, stream>>>(x, ln1_w + (size_t)l * DIM, ln1_b + (size_t)l * DIM, hb, DIM);
    gemm_bf16_wmma<1024, 1024, 0><<<gDIM, blk256, 0, stream>>>(
        hb, wq + (size_t)l * DIM * H * DH, nullptr, nullptr, qb);
    gemm_bf16_wmma<128, 1024, 0><<<gKV, blk256, 0, stream>>>(
        hb, wkv + (size_t)l * DIM * 2 * DH, nullptr, nullptr, kvb);
    build_kv_kernel<<<(B * J + 255) / 256, blk256, 0, stream>>>(
        kvb, xl_mems, knn_layer, kc, vt, l, B, N, XL);
    attention_wmma<<<B * H * (N / 16), blk32, attn_shmem, stream>>>(
        qb, kc, vt, dbkn, db_kv, knn_scale, knn_layer, topk, ao, l, B, H, N, J, XL, M);
    gemm_bf16_wmma<1024, 1024, EPI_BIAS | EPI_RESID><<<gDIM, blk256, 0, stream>>>(
        ao, wo + (size_t)l * H * DH * DIM, wo_b + (size_t)l * DIM, x, x2);
    ln_kernel<<<ROWS, blk256, 0, stream>>>(x2, ln2_w + (size_t)l * DIM, ln2_b + (size_t)l * DIM, hb, DIM);
    gemm_bf16_wmma<4096, 1024, EPI_BIAS | EPI_GELU><<<gFF, blk256, 0, stream>>>(
        hb, ff_w1 + (size_t)l * DIM * 4 * DIM, ff_b1 + (size_t)l * 4 * DIM, nullptr, ffh);
    gemm_bf16_wmma<1024, 4096, EPI_BIAS | EPI_RESID><<<gDIM, blk256, 0, stream>>>(
        ffh, ff_w2 + (size_t)l * 4 * DIM * DIM, ff_b2 + (size_t)l * DIM, x2, x);
  }

  ln_kernel<<<ROWS, blk256, 0, stream>>>(x, lnf_w, lnf_b, hb, DIM);
  gemm_bf16_wmma<32000, 1024, EPI_BIAS><<<gLG, blk256, 0, stream>>>(
      hb, w_logits, b_logits, nullptr, (float*)d_out);
}
